// Teacher_60421599920393
// MI455X (gfx1250) — matrix operand order repaired and checked by reading, not compiled, not run on hardware
//
#include <hip/hip_runtime.h>
#include <hip/hip_bf16.h>

typedef __attribute__((ext_vector_type(16))) __bf16 v16bf;
typedef __attribute__((ext_vector_type(8)))  float  v8f;
typedef __attribute__((ext_vector_type(4)))  float  v4f;
typedef __attribute__((ext_vector_type(2)))  float  v2f;

#define B_SZ 64
#define T_SZ 512
#define H_SZ 1024
#define K_SZ 64
#define START_TAG 62
#define STOP_TAG  63
#define NEGV (-10000.0f)

// ---------------------------------------------------------------------------
// Kernel 1: feats[row][k] = hiddens[row][:] @ W[:,k] + b[k]
// bf16 split (hi+lo) WMMA GEMM. W staged in LDS pre-swizzled into the
// 16-bit 32x16 B-fragment layout (hi plane + lo plane, 128KB each).
// Block: 512 threads = 16 waves, each wave owns one 16-row tile x all 64 cols.
// Grid: 128 blocks -> 2048 row tiles = 32768 rows.
// ---------------------------------------------------------------------------
__global__ __launch_bounds__(512)
void emis_gemm(const float* __restrict__ hid, const float* __restrict__ W,
               const float* __restrict__ bias, float* __restrict__ feats) {
  extern __shared__ char lds[];
  __bf16* sWhi = (__bf16*)lds;                    // 128 frags * 32 lanes * 16 el
  __bf16* sWlo = (__bf16*)(lds + 128 * 1024);

  const int tid = threadIdx.x;

  // ---- Stage W: fp32 -> (hi,lo) bf16, swizzled into WMMA B-fragment order.
  // Fragment f = kchunk*4 + ntile; within a fragment each lane holds 16
  // contiguous bf16 (32B): lane = (n%16) + (k%32>=16 ? 16:0), element = k%16.
  for (int idx = tid; idx < H_SZ * K_SZ; idx += 512) {
    const int k = idx >> 6, n = idx & 63;
    const int kin  = k & 31;
    const int frag = ((k >> 5) << 2) | (n >> 4);
    const int ln   = (n & 15) | (kin & 16);
    const int off  = frag * 512 + ln * 16 + (kin & 15);
    const float w  = W[idx];
    const __bf16 h = (__bf16)w;
    sWhi[off] = h;
    sWlo[off] = (__bf16)(w - (float)h);
  }
  __syncthreads();

  const int wv = tid >> 5, lane = tid & 31;
  const int rowbase = (blockIdx.x * 16 + wv) * 16;
  const int mrow = rowbase + (lane & 15);            // A: M = lane%16
  const int ksub = (lane & 16) ? 16 : 0;             // A halves by lane group
  const float* pA = hid + (size_t)mrow * H_SZ + ksub;

  const v16bf* fhi = (const v16bf*)sWhi;
  const v16bf* flo = (const v16bf*)sWlo;

  v8f acc[4] = {v8f{}, v8f{}, v8f{}, v8f{}};

  for (int c = 0; c < 32; ++c) {                     // K chunks of 32
    const float* p = pA + c * 32;
    v4f f0 = *(const v4f*)(p);
    v4f f1 = *(const v4f*)(p + 4);
    v4f f2 = *(const v4f*)(p + 8);
    v4f f3 = *(const v4f*)(p + 12);
    __builtin_prefetch((const void*)(p + 96), 0, 0); // global_prefetch_b8

    float fl[16];
#pragma unroll
    for (int e = 0; e < 4; ++e) {
      fl[e] = f0[e]; fl[4 + e] = f1[e]; fl[8 + e] = f2[e]; fl[12 + e] = f3[e];
    }
    v16bf ahi, alo;
#pragma unroll
    for (int e = 0; e < 16; ++e) {
      const __bf16 h = (__bf16)fl[e];
      ahi[e] = h;
      alo[e] = (__bf16)(fl[e] - (float)h);
    }

    const int fb = c * 4;
#pragma unroll
    for (int nt = 0; nt < 4; ++nt) {
      v16bf bh = fhi[(fb + nt) * 32 + lane];
      v16bf bl = flo[(fb + nt) * 32 + lane];
      acc[nt] = __builtin_amdgcn_wmma_f32_16x16x32_bf16(
          false, ahi, false, bh, (short)0, acc[nt], false, false);
      acc[nt] = __builtin_amdgcn_wmma_f32_16x16x32_bf16(
          false, alo, false, bh, (short)0, acc[nt], false, false);
      acc[nt] = __builtin_amdgcn_wmma_f32_16x16x32_bf16(
          false, ahi, false, bl, (short)0, acc[nt], false, false);
    }
  }

  // Epilogue: + bias, store fp32. C/D layout: M = r + (lane>=16?8:0), N = lane%16.
  const int hi8 = (lane & 16) ? 8 : 0;
#pragma unroll
  for (int nt = 0; nt < 4; ++nt) {
    const int ncol = nt * 16 + (lane & 15);
    const float bv = bias[ncol];
#pragma unroll
    for (int r = 0; r < 8; ++r) {
      const int row = rowbase + hi8 + r;
      feats[(size_t)row * K_SZ + ncol] = acc[nt][r] + bv;
    }
  }
}

// ---------------------------------------------------------------------------
// Kernel 2: CRF forward log-partition via exp-domain matmul scan.
//   E = exp(pinned trans)  (NEG -> exact 0)
//   per step: V = E @ U ; fv = m + log(V) + feat ; m' = colmax(fv) ;
//             U' = exp(fv - m')
// One block = 16 sequences, 4 waves; wave i owns V rows [16i,16i+16) and
// keeps its 16 f32 WMMA A-fragments of E in registers for the entire scan.
// ---------------------------------------------------------------------------
__global__ __launch_bounds__(128)
void crf_scan(const float* __restrict__ trans, const float* __restrict__ feats,
              float* __restrict__ out) {
  __shared__ float fvb[64][16];
  __shared__ float Ub[64][16];
  __shared__ float pm[4][16];
  __shared__ float Mv[16];

  const int tid  = threadIdx.x;
  const int lane = tid & 31;
  const int wv   = tid >> 5;
  const int s    = lane & 15;                 // local sequence (N dim)
  const int kg   = (lane & 16) ? 2 : 0;       // K sub-offset (32-bit frag layout)
  const int hi8  = (lane & 16) ? 8 : 0;
  const int seqbase = blockIdx.x * 16;

  // A fragments: E rows 16*wv..16*wv+15, 16 chunks of K=4.
  v2f Afr[16];
  {
    const int m = 16 * wv + (lane & 15);
#pragma unroll
    for (int c = 0; c < 16; ++c) {
#pragma unroll
      for (int v = 0; v < 2; ++v) {
        const int k = 4 * c + kg + v;
        float tv = trans[m * 64 + k];
        if (m == START_TAG) tv = NEGV;
        if (k == STOP_TAG)  tv = NEGV;
        Afr[c][v] = __expf(tv);
      }
    }
  }

  // Init: fv0 = NEG except START=0  =>  m=0, U = one-hot(START).
  for (int j = tid; j < 64 * 16; j += 128) {
    Ub[j >> 4][j & 15] = ((j >> 4) == START_TAG) ? 1.0f : 0.0f;
  }
  if (tid < 16) Mv[tid] = 0.0f;
  __syncthreads();

  const int nbase = 16 * wv + hi8;            // first of this lane's 8 tag rows
  const float* fptr =
      feats + ((size_t)(seqbase + s) * T_SZ) * K_SZ + nbase;
  v4f fA = *(const v4f*)(fptr);
  v4f fB = *(const v4f*)(fptr + 4);

  for (int t = 0; t < T_SZ; ++t) {
    // V tile = sum_c A[c] * B[c]  (fp32 WMMA, K=4 per step)
    v8f acc = {};
#pragma unroll
    for (int c = 0; c < 16; ++c) {
      v2f bf;
      bf[0] = Ub[4 * c + kg + 0][s];
      bf[1] = Ub[4 * c + kg + 1][s];
      acc = __builtin_amdgcn_wmma_f32_16x16x4_f32(
          false, Afr[c], false, bf, (short)0, acc, false, false);
    }

    // Prefetch next step's emissions (latency hidden across barriers).
    v4f nA = fA, nB = fB;
    if (t + 1 < T_SZ) {
      nA = *(const v4f*)(fptr + (size_t)(t + 1) * K_SZ);
      nB = *(const v4f*)(fptr + (size_t)(t + 1) * K_SZ + 4);
    }

    const float mprev = Mv[s];
#pragma unroll
    for (int r = 0; r < 8; ++r) {
      const float fe = (r < 4) ? fA[r] : fB[r - 4];
      fvb[nbase + r][s] = mprev + __logf(acc[r]) + fe;
    }
    fA = nA; fB = nB;
    __syncthreads();

    if (tid < 64) {                           // partial column max
      const int ss = tid & 15, q = tid >> 4;
      float m = fvb[q * 16][ss];
      for (int p = 1; p < 16; ++p) m = fmaxf(m, fvb[q * 16 + p][ss]);
      pm[q][ss] = m;
    }
    __syncthreads();
    if (tid < 16) {
      Mv[tid] = fmaxf(fmaxf(pm[0][tid], pm[1][tid]),
                      fmaxf(pm[2][tid], pm[3][tid]));
    }
    __syncthreads();
    for (int j = tid; j < 64 * 16; j += 128) { // U' = exp(fv - m')
      const int p = j >> 4, ss = j & 15;
      Ub[p][ss] = __expf(fvb[p][ss] - Mv[ss]);
    }
    __syncthreads();
  }

  // Terminal: out = m + log( sum_p exp(trans[STOP,p]) * U[p] )
  if (tid < 16) {
    float sum = 0.0f;
    for (int p = 0; p < 64; ++p) {
      float tv = trans[STOP_TAG * 64 + p];
      if (p == STOP_TAG) tv = NEGV;
      sum += __expf(tv) * Ub[p][tid];
    }
    out[seqbase + tid] = Mv[tid] + __logf(sum);
  }
}

// ---------------------------------------------------------------------------
extern "C" void kernel_launch(void* const* d_in, const int* in_sizes, int n_in,
                              void* d_out, int out_size, void* d_ws, size_t ws_size,
                              hipStream_t stream) {
  const float* hid   = (const float*)d_in[0];   // [64,512,1024]
  const float* W     = (const float*)d_in[1];   // [1024,64]
  const float* bias  = (const float*)d_in[2];   // [64]
  const float* trans = (const float*)d_in[3];   // [64,64]
  float* out   = (float*)d_out;                 // [64]
  float* feats = (float*)d_ws;                  // 32768*64 fp32 = 8MB scratch

  (void)in_sizes; (void)n_in; (void)out_size; (void)ws_size;

  const size_t gemm_lds = 256 * 1024;           // W hi+lo fragment planes
  hipFuncSetAttribute((const void*)emis_gemm,
                      hipFuncAttributeMaxDynamicSharedMemorySize,
                      (int)gemm_lds);

  hipLaunchKernelGGL(emis_gemm, dim3(128), dim3(512), gemm_lds, stream,
                     hid, W, bias, feats);
  hipLaunchKernelGGL(crf_scan, dim3(4), dim3(128), 0, stream,
                     trans, feats, out);
}
